// s4MambaModule_64330020159661
// MI455X (gfx1250) — compile-verified, matrix operand order checked by reading, not run on hardware
//
#include <hip/hip_runtime.h>
#include <hip/hip_bf16.h>

// ---------------------------------------------------------------------------
// Problem constants (from reference)
// ---------------------------------------------------------------------------
#define BB   4
#define LL   2048
#define DM   1024
#define NST  16
#define WW   4
#define DIN  2048          // EXPAND * DM
#define E2   (2 * DIN)     // 4096, in_proj output channels
#define MM   (BB * LL)     // 8192 GEMM rows

typedef __attribute__((ext_vector_type(16))) __bf16 v16bf;
typedef __attribute__((ext_vector_type(8)))  __bf16 v8bf;
typedef __attribute__((ext_vector_type(8)))  float  v8f;

// LDS tile geometry: 128 rows x 32 K-elements, padded to 40 to keep 16B
// alignment of every fragment chunk (row stride 80B) and spread banks.
#define TROWS 128
#define TK    32
#define TPAD  40

// ---------------------------------------------------------------------------
// Kernel 1: f32 -> bf16 conversion (grid-stride)
// ---------------------------------------------------------------------------
__global__ __launch_bounds__(256) void cvt_f32_bf16(const float* __restrict__ in,
                                                    __bf16* __restrict__ out,
                                                    size_t n) {
    size_t i = (size_t)blockIdx.x * blockDim.x + threadIdx.x;
    size_t stride = (size_t)gridDim.x * blockDim.x;
    for (; i < n; i += stride) out[i] = (__bf16)in[i];
}

// ---------------------------------------------------------------------------
// CDNA5 async global->LDS copy (16B per lane), tracked by ASYNCcnt.
// ISA: GLOBAL_LOAD_ASYNC_TO_LDS_B128  vdst(LDS byte addr), vaddr(64b), off
// ---------------------------------------------------------------------------
__device__ inline void async_cp16(const __bf16* g, __bf16* l) {
    unsigned lds = (unsigned)(uintptr_t)(__attribute__((address_space(3))) __bf16*)l;
    asm volatile("global_load_async_to_lds_b128 %0, %1, off"
                 :: "v"(lds), "v"(g) : "memory");
}
__device__ inline void wait_async0() {
    asm volatile("s_wait_asynccnt 0x0" ::: "memory");
}

// ---------------------------------------------------------------------------
// Fragment assembly from an LDS row (16-bit A/B WMMA layout,
// CDNA5 ISA): lanes 0-15 hold K octets {0..7,16..23},
// lanes 16-31 hold {8..15,24..31}.  Two 16B ds_load_b128 per fragment.
// ---------------------------------------------------------------------------
__device__ inline v16bf frag_row(const __bf16* rowPtr, int half) {
    const v8bf lo = *(const v8bf*)(rowPtr + half * 8);
    const v8bf hi = *(const v8bf*)(rowPtr + 16 + half * 8);
    return __builtin_shufflevector(lo, hi,
               0,1,2,3,4,5,6,7,8,9,10,11,12,13,14,15);
}

// ---------------------------------------------------------------------------
// Kernels 2/4: BF16 WMMA GEMM, C[M,N] = A[M,K] (row-major) x W[N,K]
// (row-major = B^T, so both tiles load contiguously along K).
//
// Block: 256 threads = 8 waves, block tile 128(M) x 128(N), K-step 32.
// LDS double buffer: A-slice + B-slice (128x32 bf16 each, padded) ~40KB of
// the 320KB WGP LDS, filled by GLOBAL_LOAD_ASYNC_TO_LDS_B128 (no VGPR
// staging, ASYNCcnt-tracked).  Waves arranged 4(M) x 2(N); each wave
// computes a 32x64 sub-tile: 2 A-frags x 4 B-frags -> 8 independent WMMA
// chains per K-step.  One barrier per K-iteration:
//   issue async(k+1 -> nxt) ; compute(cur) ; s_wait_asynccnt 0 ; barrier
// ---------------------------------------------------------------------------
__global__ __launch_bounds__(256) void wmma_gemm_bf16(const __bf16* __restrict__ A,
                                                      const __bf16* __restrict__ Wt,
                                                      float* __restrict__ C,
                                                      int M, int N, int K) {
    __shared__ __bf16 Ash[2][TROWS][TPAD];
    __shared__ __bf16 Bsh[2][TROWS][TPAD];

    const int tid  = threadIdx.x;
    const int wave = tid >> 5;
    const int lane = tid & 31;
    const int r    = lane & 15;      // row (A-frag) / col (B-frag) / n (C)
    const int half = lane >> 4;      // K-octet selector

    const int wm = wave >> 1;        // 0..3 -> M offset wm*32
    const int wn = wave & 1;         // 0..1 -> N offset wn*64

    const int mBase = blockIdx.x * 128;
    const int nBase = blockIdx.y * 128;

    // Staging map: tile = 128 rows x 32 cols = 512 chunks of 8 bf16 (16B).
    // Thread t DMAs chunks t and t+256 for both A and B slices.
    const int c0   = tid;
    const int c1   = tid + 256;
    const int r0   = c0 >> 2, s0 = (c0 & 3) * 8;
    const int r1   = c1 >> 2, s1 = (c1 & 3) * 8;

    const __bf16* aG0 = A  + (size_t)(mBase + r0) * K + s0;
    const __bf16* aG1 = A  + (size_t)(mBase + r1) * K + s1;
    const __bf16* bG0 = Wt + (size_t)(nBase + r0) * K + s0;
    const __bf16* bG1 = Wt + (size_t)(nBase + r1) * K + s1;

    v8f acc[2][4];
#pragma unroll
    for (int i = 0; i < 2; ++i)
#pragma unroll
        for (int j = 0; j < 4; ++j) acc[i][j] = (v8f){};

    const int nk = K / TK;

    // --- stage K-slice 0 into buffer 0 via async DMA ---
    async_cp16(aG0, &Ash[0][r0][s0]);
    async_cp16(aG1, &Ash[0][r1][s1]);
    async_cp16(bG0, &Bsh[0][r0][s0]);
    async_cp16(bG1, &Bsh[0][r1][s1]);
    wait_async0();
    __syncthreads();

    for (int kt = 0; kt < nk; ++kt) {
        const int cur = kt & 1;

        // issue async DMA of next K-slice into the other buffer; all readers
        // of that buffer passed the barrier that ended iteration kt-1.
        if (kt + 1 < nk) {
            const int ko  = (kt + 1) * TK;
            const int nxt = cur ^ 1;
            async_cp16(aG0 + ko, &Ash[nxt][r0][s0]);
            async_cp16(aG1 + ko, &Ash[nxt][r1][s1]);
            async_cp16(bG0 + ko, &Bsh[nxt][r0][s0]);
            async_cp16(bG1 + ko, &Bsh[nxt][r1][s1]);
        }

        // ---- compute from LDS buffer `cur` ----
        const __bf16 (*At)[TPAD] = Ash[cur];
        const __bf16 (*Bt)[TPAD] = Bsh[cur];

        const v16bf a0 = frag_row(&At[wm * 32 +  0 + r][0], half);
        const v16bf a1 = frag_row(&At[wm * 32 + 16 + r][0], half);
        const v16bf b0 = frag_row(&Bt[wn * 64 +  0 + r][0], half);
        const v16bf b1 = frag_row(&Bt[wn * 64 + 16 + r][0], half);
        const v16bf b2 = frag_row(&Bt[wn * 64 + 32 + r][0], half);
        const v16bf b3 = frag_row(&Bt[wn * 64 + 48 + r][0], half);

        acc[0][0] = __builtin_amdgcn_wmma_f32_16x16x32_bf16(false, a0, false, b0,
                        (short)0, acc[0][0], false, false);
        acc[0][1] = __builtin_amdgcn_wmma_f32_16x16x32_bf16(false, a0, false, b1,
                        (short)0, acc[0][1], false, false);
        acc[0][2] = __builtin_amdgcn_wmma_f32_16x16x32_bf16(false, a0, false, b2,
                        (short)0, acc[0][2], false, false);
        acc[0][3] = __builtin_amdgcn_wmma_f32_16x16x32_bf16(false, a0, false, b3,
                        (short)0, acc[0][3], false, false);
        acc[1][0] = __builtin_amdgcn_wmma_f32_16x16x32_bf16(false, a1, false, b0,
                        (short)0, acc[1][0], false, false);
        acc[1][1] = __builtin_amdgcn_wmma_f32_16x16x32_bf16(false, a1, false, b1,
                        (short)0, acc[1][1], false, false);
        acc[1][2] = __builtin_amdgcn_wmma_f32_16x16x32_bf16(false, a1, false, b2,
                        (short)0, acc[1][2], false, false);
        acc[1][3] = __builtin_amdgcn_wmma_f32_16x16x32_bf16(false, a1, false, b3,
                        (short)0, acc[1][3], false, false);

        // ---- close the pipeline stage: my DMAs landed, everyone done ----
        if (kt + 1 < nk) {
            wait_async0();
            __syncthreads();
        }
    }

    // ---- store C: lane holds n = r; VGPR i holds m = half*8 + i ----
#pragma unroll
    for (int i = 0; i < 2; ++i) {
        float* cRow = C + (size_t)(mBase + wm * 32 + i * 16 + half * 8) * N
                        + nBase + wn * 64 + r;
#pragma unroll
        for (int j = 0; j < 4; ++j) {
#pragma unroll
            for (int e = 0; e < 8; ++e)
                cRow[(size_t)e * N + j * 16] = acc[i][j][e];
        }
    }
}

// ---------------------------------------------------------------------------
// Kernel 3: fused causal depthwise conv(W=4) + SiLU + diagonal SSM recurrence
// + D-skip + SiLU(z) gating. One thread per (b, h) channel, coalesced over h.
// xz layout from GEMM1: (b, l, e) with e in [0, 4096); x = e<DIN, z = e>=DIN.
// Recurrence equivalent of the reference FFT conv:
//   s_t = exp(dt*A) * s_{t-1} + u_t ;  y_t = Re(2*C*dB . s_t) + D*u_t
// ---------------------------------------------------------------------------
__global__ __launch_bounds__(256) void conv_ssm_gate(const float* __restrict__ xz,
                                                     const float* __restrict__ conv_w,
                                                     const float* __restrict__ conv_b,
                                                     const float* __restrict__ log_dt,
                                                     const float* __restrict__ A_re,
                                                     const float* __restrict__ A_im,
                                                     const float* __restrict__ C_re,
                                                     const float* __restrict__ C_im,
                                                     const float* __restrict__ D_skip,
                                                     __bf16* __restrict__ gated) {
    const int tid = blockIdx.x * blockDim.x + threadIdx.x;
    if (tid >= BB * DIN) return;
    const int h = tid % DIN;
    const int b = tid / DIN;

    const float dt = __expf(log_dt[h]);

    float lam_re[NST], lam_im[NST], co_re[NST], co_im[NST];
#pragma unroll
    for (int n = 0; n < NST; ++n) {
        const float ar = A_re[h * NST + n];
        const float ai = A_im[h * NST + n];
        const float er = __expf(dt * ar);
        float c, s;
        __sincosf(dt * ai, &s, &c);
        const float e_re = er * c;           // exp(dt*A)
        const float e_im = er * s;
        lam_re[n] = e_re;
        lam_im[n] = e_im;
        // dB = (exp(dt*A) - 1) / A   (complex divide)
        const float nr = e_re - 1.0f, ni = e_im;
        const float den = 1.0f / (ar * ar + ai * ai);
        const float dB_re = (nr * ar + ni * ai) * den;
        const float dB_im = (ni * ar - nr * ai) * den;
        // coef = 2 * C * dB
        const float cr = C_re[h * NST + n], ci = C_im[h * NST + n];
        co_re[n] = 2.0f * (cr * dB_re - ci * dB_im);
        co_im[n] = 2.0f * (cr * dB_im + ci * dB_re);
    }

    const float w0 = conv_w[h * WW + 0], w1 = conv_w[h * WW + 1];
    const float w2 = conv_w[h * WW + 2], w3 = conv_w[h * WW + 3];
    const float cb = conv_b[h];
    const float dsk = D_skip[h];

    const float* xb = xz + (size_t)b * LL * E2;   // (l, e) plane

    float xm3 = 0.f, xm2 = 0.f, xm1 = 0.f;
    float s_re[NST], s_im[NST];
#pragma unroll
    for (int n = 0; n < NST; ++n) { s_re[n] = 0.f; s_im[n] = 0.f; }

    __bf16* gb = gated + (size_t)b * LL * DIN + h;

    for (int t = 0; t < LL; ++t) {
        const float x0 = xb[(size_t)t * E2 + h];
        float u = w0 * xm3 + w1 * xm2 + w2 * xm1 + w3 * x0 + cb;
        xm3 = xm2; xm2 = xm1; xm1 = x0;
        u = u * (1.0f / (1.0f + __expf(-u)));        // SiLU

        float y = dsk * u;
#pragma unroll
        for (int n = 0; n < NST; ++n) {
            const float sr = lam_re[n] * s_re[n] - lam_im[n] * s_im[n] + u;
            const float si = lam_re[n] * s_im[n] + lam_im[n] * s_re[n];
            s_re[n] = sr; s_im[n] = si;
            y += co_re[n] * sr - co_im[n] * si;
        }

        const float zv = xb[(size_t)t * E2 + DIN + h];
        const float g = zv * (1.0f / (1.0f + __expf(-zv)));
        gb[(size_t)t * DIN] = (__bf16)(y * g);
    }
}

// ---------------------------------------------------------------------------
// Host launcher
// ---------------------------------------------------------------------------
static inline size_t align256(size_t x) { return (x + 255) & ~(size_t)255; }

extern "C" void kernel_launch(void* const* d_in, const int* in_sizes, int n_in,
                              void* d_out, int out_size, void* d_ws, size_t ws_size,
                              hipStream_t stream) {
    const float* hs    = (const float*)d_in[0];   // (B, L, DM)
    const float* w_in  = (const float*)d_in[1];   // (2*DIN, DM)
    const float* cw    = (const float*)d_in[2];   // (DIN, W)
    const float* cbias = (const float*)d_in[3];   // (DIN,)
    const float* w_out = (const float*)d_in[4];   // (DM, DIN)
    const float* logdt = (const float*)d_in[5];
    const float* a_re  = (const float*)d_in[6];
    const float* a_im  = (const float*)d_in[7];
    const float* c_re  = (const float*)d_in[8];
    const float* c_im  = (const float*)d_in[9];
    const float* dskip = (const float*)d_in[10];

    // Workspace carve-up
    char* ws = (char*)d_ws;
    size_t off = 0;
    __bf16* hsb = (__bf16*)(ws + off); off = align256(off + (size_t)MM * DM * sizeof(__bf16));
    __bf16* w1b = (__bf16*)(ws + off); off = align256(off + (size_t)E2 * DM * sizeof(__bf16));
    __bf16* w2b = (__bf16*)(ws + off); off = align256(off + (size_t)DM * DIN * sizeof(__bf16));
    float*  xzf = (float*) (ws + off); off = align256(off + (size_t)MM * E2 * sizeof(float));
    __bf16* gtb = (__bf16*)(ws + off); off = align256(off + (size_t)MM * DIN * sizeof(__bf16));
    (void)ws_size;

    // 1) f32 -> bf16 conversions
    cvt_f32_bf16<<<2048, 256, 0, stream>>>(hs,    hsb, (size_t)MM * DM);
    cvt_f32_bf16<<<1024, 256, 0, stream>>>(w_in,  w1b, (size_t)E2 * DM);
    cvt_f32_bf16<<<512,  256, 0, stream>>>(w_out, w2b, (size_t)DM * DIN);

    // 2) in_proj: xz[M, 4096] = hs[M, 1024] x w_in[4096, 1024]^T
    {
        dim3 grid(MM / 128, E2 / 128);
        wmma_gemm_bf16<<<grid, 256, 0, stream>>>(hsb, w1b, xzf, MM, E2, DM);
    }

    // 3) fused conv + SiLU + SSM scan + gate  -> gated bf16 (b, l, h)
    conv_ssm_gate<<<(BB * DIN + 255) / 256, 256, 0, stream>>>(
        xzf, cw, cbias, logdt, a_re, a_im, c_re, c_im, dskip, gtb);

    // 4) out_proj: out[M, 1024] = gated[M, 2048] x w_out[1024, 2048]^T
    {
        dim3 grid(MM / 128, DM / 128);
        wmma_gemm_bf16<<<grid, 256, 0, stream>>>(gtb, w2b, (float*)d_out, MM, DM, DIN);
    }
}
